// Transformer_28054726377955
// MI455X (gfx1250) — matrix operand order repaired and checked by reading, not compiled, not run on hardware
//
#include <hip/hip_runtime.h>
#include <hip/hip_bf16.h>
#include <math.h>

// ---------------------------------------------------------------------------
// Model constants (match reference)
// ---------------------------------------------------------------------------
#define D_MODEL   1024
#define NUM_LAYERS 4
#define VOCAB     32000
#define NUM_HEADS 8
#define HEAD_DIM  128
#define FFN_DIM   2048
#define BATCH     4
#define SEQ       1024
#define NTOK      (BATCH * SEQ)   // 4096 rows

typedef __bf16 v8bf  __attribute__((ext_vector_type(8)));
typedef __bf16 v16bf __attribute__((ext_vector_type(16)));
typedef float  v8f   __attribute__((ext_vector_type(8)));

// ---------------------------------------------------------------------------
// WMMA fragment loaders (wave32 layouts per CDNA5 ISA 7.12.2)
//
// A (16x32 bf16): lane = tid&31, row M = lane&15.
//   element e<8  -> K = (lane>>4)*8 + e
//   element e>=8 -> K = 16 + (lane>>4)*8 + (e-8)
// => two 16B chunks at k0+(lane>>4)*8 and k0+16+(lane>>4)*8.
//
// B (32x16 bf16) from row-major-by-N storage Bt[N][K]:
//   N = lane&15, element e<8 -> K = (lane>>4)*8 + e,
//   element e>=8 -> K = 16 + (lane>>4)*8 + (e-8)  (same as A)
// => two 16B chunks at k0+(lane>>4)*8 and k0+16+(lane>>4)*8.
// ---------------------------------------------------------------------------
__device__ __forceinline__ v16bf load_a_frag(const __bf16* A, int lda, int m0,
                                             int k0, int lane) {
  const __bf16* p = A + (size_t)(m0 + (lane & 15)) * lda + k0 + ((lane >> 4) << 3);
  v8bf lo = *(const v8bf*)p;
  v8bf hi = *(const v8bf*)(p + 16);
  return __builtin_shufflevector(lo, hi, 0, 1, 2, 3, 4, 5, 6, 7,
                                 8, 9, 10, 11, 12, 13, 14, 15);
}

__device__ __forceinline__ v16bf load_b_row(const __bf16* Bt, int ldk, int n0,
                                            int k0, int lane) {
  const __bf16* p = Bt + (size_t)(n0 + (lane & 15)) * ldk + k0 + ((lane >> 4) << 3);
  v8bf lo = *(const v8bf*)p;   // two 16B chunks, 16B aligned at all call sites
  v8bf hi = *(const v8bf*)(p + 16);
  return __builtin_shufflevector(lo, hi, 0, 1, 2, 3, 4, 5, 6, 7,
                                 8, 9, 10, 11, 12, 13, 14, 15);
}

// ---------------------------------------------------------------------------
// Weight prep: fp32 [K,N] row-major -> bf16 [N,K] row-major ("transposed")
// ---------------------------------------------------------------------------
__global__ void k_cvt_t(const float* __restrict__ W, __bf16* __restrict__ Wt,
                        int K, int N) {
  size_t o = (size_t)blockIdx.x * 256 + threadIdx.x;
  size_t total = (size_t)K * N;
  if (o >= total) return;
  size_t n = o / K, k = o % K;
  Wt[o] = (__bf16)W[k * N + n];
}

// ---------------------------------------------------------------------------
// Embedding gather: x[n,:] = emb[tokens[n],:]
// ---------------------------------------------------------------------------
__global__ void k_embed(const int* __restrict__ tokens,
                        const float* __restrict__ emb, float* __restrict__ x) {
  int n = blockIdx.x;
  int t = tokens[n];
  for (int d = threadIdx.x; d < D_MODEL; d += blockDim.x)
    x[(size_t)n * D_MODEL + d] = emb[(size_t)t * D_MODEL + d];
}

// ---------------------------------------------------------------------------
// LayerNorm (fp32 in -> bf16 out), one 256-thread block per row
// ---------------------------------------------------------------------------
__global__ void k_layernorm(const float* __restrict__ x,
                            const float* __restrict__ g,
                            const float* __restrict__ b,
                            __bf16* __restrict__ y) {
  int row = blockIdx.x;
  const float* xr = x + (size_t)row * D_MODEL;
  float s = 0.f, s2 = 0.f;
  for (int i = threadIdx.x; i < D_MODEL; i += blockDim.x) {
    float v = xr[i]; s += v; s2 += v * v;
  }
#pragma unroll
  for (int m = 16; m >= 1; m >>= 1) {
    s  += __shfl_xor(s, m, 32);
    s2 += __shfl_xor(s2, m, 32);
  }
  __shared__ float shs[8], shs2[8], stats[2];
  int wid = threadIdx.x >> 5, lane = threadIdx.x & 31;
  if (lane == 0) { shs[wid] = s; shs2[wid] = s2; }
  __syncthreads();
  if (threadIdx.x == 0) {
    float ts = 0.f, ts2 = 0.f;
#pragma unroll
    for (int i = 0; i < 8; ++i) { ts += shs[i]; ts2 += shs2[i]; }
    float mu = ts / D_MODEL;
    float var = ts2 / D_MODEL - mu * mu;
    stats[0] = mu;
    stats[1] = rsqrtf(var + 1e-5f);
  }
  __syncthreads();
  float mu = stats[0], inv = stats[1];
  for (int i = threadIdx.x; i < D_MODEL; i += blockDim.x)
    y[(size_t)row * D_MODEL + i] = (__bf16)((xr[i] - mu) * inv * g[i] + b[i]);
}

// ---------------------------------------------------------------------------
// Generic bf16 WMMA GEMM:  out = act(A[M,K] @ Bt[N,K]^T + bias (+ resid))
// 128 threads = 4 waves; block tile 64x128; Bt tile (128x32, 8KB) staged in
// LDS shared by all 4 waves; each wave: 16 rows x 128 cols = 8 accumulators,
// 8 WMMAs per 32-wide K chunk with a single A fragment.
// ---------------------------------------------------------------------------
template <bool GELU_, bool RESID, bool OUTBF>
__global__ void __launch_bounds__(128, 1)
k_gemm(const __bf16* __restrict__ A, const __bf16* __restrict__ Bt,
       const float* __restrict__ bias, const float* __restrict__ resid,
       void* __restrict__ outp, int M, int N, int K) {
  __shared__ __bf16 sB[128 * 32];   // [n_local][k_local], 8 KB
  int wid = threadIdx.x >> 5;
  int lane = threadIdx.x & 31;
  int n0 = blockIdx.x * 128;
  int m0 = blockIdx.y * 64 + wid * 16;

  v8f acc[8] = {};

  for (int k0 = 0; k0 < K; k0 += 32) {
    // cooperative stage: thread t copies one full 64-byte row (4 x 16B)
    {
      int row = threadIdx.x;                // 0..127
      const __bf16* src = Bt + (size_t)(n0 + row) * K + k0;
#pragma unroll
      for (int q = 0; q < 4; ++q)
        *(v8bf*)(&sB[row * 32 + q * 8]) = *(const v8bf*)(src + q * 8);
      if (k0 + 32 < K)
        __builtin_prefetch(src + 32, 0, 1);   // global_prefetch_b8 next tile
    }
    __syncthreads();

    v16bf a = load_a_frag(A, K, m0, k0, lane);
#pragma unroll
    for (int s = 0; s < 8; ++s) {
      v16bf bfrag = load_b_row(sB, 32, s * 16, 0, lane);
      acc[s] = __builtin_amdgcn_wmma_f32_16x16x32_bf16(
          false, a, false, bfrag, (short)0, acc[s], false, false);
    }
    __syncthreads();
  }

  // epilogue (C layout: element r -> row m0 + r + (lane>=16)*8, col lane&15)
#pragma unroll
  for (int s = 0; s < 8; ++s) {
#pragma unroll
    for (int r = 0; r < 8; ++r) {
      int row = m0 + r + ((lane >> 4) << 3);
      int col = n0 + s * 16 + (lane & 15);
      float v = acc[s][r] + bias[col];
      if (GELU_) v = 0.5f * v * (1.0f + erff(v * 0.70710678118f));
      if (RESID) v += resid[(size_t)row * N + col];
      if (OUTBF) ((__bf16*)outp)[(size_t)row * N + col] = (__bf16)v;
      else       ((float*)outp)[(size_t)row * N + col] = v;
    }
  }
}

// ---------------------------------------------------------------------------
// Split qkv, apply RoPE to q/k, write Q,K [B,H,S,Dh] and V transposed
// [B,H,Dh,S] (so ctx B-fragments are contiguous in the key dimension).
// grid (S, B*H), 128 threads = one per channel.
// ---------------------------------------------------------------------------
__global__ void k_rope(const __bf16* __restrict__ qkv, __bf16* __restrict__ Q,
                       __bf16* __restrict__ Kc, __bf16* __restrict__ Vt) {
  int s = blockIdx.x;
  int bh = blockIdx.y;
  int c = threadIdx.x;  // 0..127
  size_t n = (size_t)(bh >> 3) * SEQ + s;
  const __bf16* base = qkv + n * (3 * D_MODEL) + (bh & 7) * HEAD_DIM;

  float q = (float)base[c];
  float k = (float)base[D_MODEL + c];
  float v = (float)base[2 * D_MODEL + c];

  int i = c & 63;
  float ang = (float)s * __powf(10000.0f, -(float)(2 * i) / 128.0f);
  float cs = __cosf(ang), sn = __sinf(ang);
  float qr = (c < 64) ? -(float)base[c + 64] : (float)base[c - 64];
  float kr = (c < 64) ? -(float)base[D_MODEL + c + 64]
                      : (float)base[D_MODEL + c - 64];
  float qo = q * cs + qr * sn;
  float ko = k * cs + kr * sn;

  size_t hd = ((size_t)bh * SEQ + s) * HEAD_DIM + c;
  Q[hd]  = (__bf16)qo;
  Kc[hd] = (__bf16)ko;
  Vt[((size_t)bh * HEAD_DIM + c) * SEQ + s] = (__bf16)v;
}

// ---------------------------------------------------------------------------
// Flash-style causal attention. grid (S/64, B*H), 128 threads = 4 waves.
// Each wave: 16 query rows, Dh=128. Scores via 8 WMMAs per 32-key tile,
// online softmax with half-wave shuffle reductions, P bounced through LDS
// to re-layout C->A fragment, ctx via 8 WMMAs. Uniform tile count -> legal
// __syncthreads. launch_bounds(…,1) gives the allocator the full VGPR file
// so aq[4]+acc[8] stay resident (no scratch spills).
// ---------------------------------------------------------------------------
__global__ void __launch_bounds__(128, 1)
k_attn(const __bf16* __restrict__ Qm, const __bf16* __restrict__ Km,
       const __bf16* __restrict__ Vm, __bf16* __restrict__ ctx) {
  __shared__ __bf16 P[4][16 * 32];
  int wid = threadIdx.x >> 5, lane = threadIdx.x & 31;
  int bh = blockIdx.y;
  int b = bh >> 3, h = bh & 7;
  int q0 = blockIdx.x * 64 + wid * 16;

  const __bf16* Qb = Qm + (size_t)bh * SEQ * HEAD_DIM;
  const __bf16* Kb = Km + (size_t)bh * SEQ * HEAD_DIM;
  const __bf16* Vb = Vm + (size_t)bh * HEAD_DIM * SEQ;

  v16bf aq[4];
#pragma unroll
  for (int c = 0; c < 4; ++c) aq[c] = load_a_frag(Qb, HEAD_DIM, q0, c * 32, lane);

  v8f acc[8] = {};
  float mi[8], li[8];
#pragma unroll
  for (int r = 0; r < 8; ++r) { mi[r] = -1e30f; li[r] = 0.f; }
  const float scale = 0.08838834764831845f;  // 1/sqrt(128)

  int ntiles = blockIdx.x * 2 + 2;           // uniform across the block
  int col = lane & 15;
  int rbase = (lane >> 4) << 3;

  for (int jt = 0; jt < ntiles; ++jt) {
    int kb = jt * 32;
    v8f s0 = {}, s1 = {};
#pragma unroll
    for (int c = 0; c < 4; ++c) {
      v16bf b0 = load_b_row(Kb, HEAD_DIM, kb,      c * 32, lane);
      v16bf b1 = load_b_row(Kb, HEAD_DIM, kb + 16, c * 32, lane);
      s0 = __builtin_amdgcn_wmma_f32_16x16x32_bf16(false, aq[c], false, b0,
                                                   (short)0, s0, false, false);
      s1 = __builtin_amdgcn_wmma_f32_16x16x32_bf16(false, aq[c], false, b1,
                                                   (short)0, s1, false, false);
    }

#pragma unroll
    for (int r = 0; r < 8; ++r) {
      int qrow = q0 + r + rbase;
      float e0 = s0[r] * scale; if (kb + col      > qrow) e0 = -1e30f;
      float e1 = s1[r] * scale; if (kb + 16 + col > qrow) e1 = -1e30f;
      float mx = fmaxf(e0, e1);
#pragma unroll
      for (int m = 8; m >= 1; m >>= 1) mx = fmaxf(mx, __shfl_xor(mx, m, 32));
      float mnew = fmaxf(mi[r], mx);
      float p0 = __expf(e0 - mnew);
      float p1 = __expf(e1 - mnew);
      float corr = __expf(mi[r] - mnew);
      float ps = p0 + p1;
#pragma unroll
      for (int m = 8; m >= 1; m >>= 1) ps += __shfl_xor(ps, m, 32);
      li[r] = li[r] * corr + ps;
      mi[r] = mnew;
#pragma unroll
      for (int t = 0; t < 8; ++t) acc[t][r] *= corr;
      int prow = r + rbase;
      P[wid][prow * 32 + col]      = (__bf16)p0;
      P[wid][prow * 32 + 16 + col] = (__bf16)p1;
    }
    __syncthreads();

    v16bf ap = load_a_frag(&P[wid][0], 32, 0, 0, lane);
#pragma unroll
    for (int t = 0; t < 8; ++t) {
      v16bf bv = load_b_row(Vb, SEQ, t * 16, kb, lane);
      acc[t] = __builtin_amdgcn_wmma_f32_16x16x32_bf16(false, ap, false, bv,
                                                       (short)0, acc[t], false, false);
    }
    __syncthreads();
  }

#pragma unroll
  for (int r = 0; r < 8; ++r) {
    int qrow = q0 + r + rbase;
    float inv = 1.0f / li[r];
    size_t base = ((size_t)b * SEQ + qrow) * D_MODEL + h * HEAD_DIM;
#pragma unroll
    for (int t = 0; t < 8; ++t)
      ctx[base + t * 16 + col] = (__bf16)(acc[t][r] * inv);
  }
}

// ---------------------------------------------------------------------------
// Host orchestration
// ---------------------------------------------------------------------------
extern "C" void kernel_launch(void* const* d_in, const int* in_sizes, int n_in,
                              void* d_out, int out_size, void* d_ws, size_t ws_size,
                              hipStream_t stream) {
  (void)in_sizes; (void)n_in; (void)out_size; (void)ws_size;

  const int*   tokens = (const int*)d_in[0];
  const float* emb    = (const float*)d_in[1];
  const float* Wqkv   = (const float*)d_in[2];
  const float* bqkv   = (const float*)d_in[3];
  const float* Wo     = (const float*)d_in[4];
  const float* bo     = (const float*)d_in[5];
  const float* ln1g   = (const float*)d_in[6];
  const float* ln1b   = (const float*)d_in[7];
  const float* ln2g   = (const float*)d_in[8];
  const float* ln2b   = (const float*)d_in[9];
  const float* W1     = (const float*)d_in[10];
  const float* b1     = (const float*)d_in[11];
  const float* W2     = (const float*)d_in[12];
  const float* b2     = (const float*)d_in[13];
  const float* lnfg   = (const float*)d_in[14];
  const float* lnfb   = (const float*)d_in[15];
  const float* Wout   = (const float*)d_in[16];
  const float* bout   = (const float*)d_in[17];
  float* out = (float*)d_out;

  char* w = (char*)d_ws;
  auto alloc = [&](size_t bytes) -> char* {
    char* p = w;
    w += (bytes + 255) & ~(size_t)255;
    return p;
  };

  __bf16* WqkvT = (__bf16*)alloc((size_t)NUM_LAYERS * 3 * D_MODEL * D_MODEL * 2);
  __bf16* WoT   = (__bf16*)alloc((size_t)NUM_LAYERS * D_MODEL * D_MODEL * 2);
  __bf16* W1T   = (__bf16*)alloc((size_t)NUM_LAYERS * D_MODEL * FFN_DIM * 2);
  __bf16* W2T   = (__bf16*)alloc((size_t)NUM_LAYERS * FFN_DIM * D_MODEL * 2);
  __bf16* WoutT = (__bf16*)alloc((size_t)VOCAB * D_MODEL * 2);
  float*  x     = (float*) alloc((size_t)NTOK * D_MODEL * 4);
  __bf16* y     = (__bf16*)alloc((size_t)NTOK * D_MODEL * 2);
  __bf16* qkv   = (__bf16*)alloc((size_t)NTOK * 3 * D_MODEL * 2);
  __bf16* Qb    = (__bf16*)alloc((size_t)NTOK * D_MODEL * 2);
  __bf16* Kb    = (__bf16*)alloc((size_t)NTOK * D_MODEL * 2);
  __bf16* Vt    = (__bf16*)alloc((size_t)NTOK * D_MODEL * 2);
  __bf16* ctx   = (__bf16*)alloc((size_t)NTOK * D_MODEL * 2);
  __bf16* h1    = (__bf16*)alloc((size_t)NTOK * FFN_DIM * 2);

  auto cvt = [&](const float* src, __bf16* dst, int K, int N) {
    size_t total = (size_t)K * N;
    int blocks = (int)((total + 255) / 256);
    k_cvt_t<<<blocks, 256, 0, stream>>>(src, dst, K, N);
  };

  // ---- weight prep (fp32 -> bf16, transposed to [N,K]) ----
  for (int l = 0; l < NUM_LAYERS; ++l) {
    cvt(Wqkv + (size_t)l * D_MODEL * 3 * D_MODEL,
        WqkvT + (size_t)l * 3 * D_MODEL * D_MODEL, D_MODEL, 3 * D_MODEL);
    cvt(Wo + (size_t)l * D_MODEL * D_MODEL,
        WoT + (size_t)l * D_MODEL * D_MODEL, D_MODEL, D_MODEL);
    cvt(W1 + (size_t)l * D_MODEL * FFN_DIM,
        W1T + (size_t)l * FFN_DIM * D_MODEL, D_MODEL, FFN_DIM);
    cvt(W2 + (size_t)l * FFN_DIM * D_MODEL,
        W2T + (size_t)l * D_MODEL * FFN_DIM, FFN_DIM, D_MODEL);
  }
  cvt(Wout, WoutT, D_MODEL, VOCAB);

  // ---- embedding ----
  k_embed<<<NTOK, 256, 0, stream>>>(tokens, emb, x);

  // ---- layers ----
  for (int l = 0; l < NUM_LAYERS; ++l) {
    k_layernorm<<<NTOK, 256, 0, stream>>>(x, ln1g + l * D_MODEL,
                                          ln1b + l * D_MODEL, y);
    // qkv = y @ Wqkv + bqkv   (out bf16 [NTOK, 3D])
    k_gemm<false, false, true><<<dim3(3 * D_MODEL / 128, NTOK / 64), 128, 0, stream>>>(
        y, WqkvT + (size_t)l * 3 * D_MODEL * D_MODEL, bqkv + (size_t)l * 3 * D_MODEL,
        nullptr, qkv, NTOK, 3 * D_MODEL, D_MODEL);
    // RoPE + reshape to [B,H,S,Dh] and Vt [B,H,Dh,S]
    k_rope<<<dim3(SEQ, BATCH * NUM_HEADS), HEAD_DIM, 0, stream>>>(qkv, Qb, Kb, Vt);
    // flash attention
    k_attn<<<dim3(SEQ / 64, BATCH * NUM_HEADS), 128, 0, stream>>>(Qb, Kb, Vt, ctx);
    // x = x + ctx @ Wo + bo
    k_gemm<false, true, false><<<dim3(D_MODEL / 128, NTOK / 64), 128, 0, stream>>>(
        ctx, WoT + (size_t)l * D_MODEL * D_MODEL, bo + (size_t)l * D_MODEL,
        x, x, NTOK, D_MODEL, D_MODEL);
    // FFN
    k_layernorm<<<NTOK, 256, 0, stream>>>(x, ln2g + l * D_MODEL,
                                          ln2b + l * D_MODEL, y);
    k_gemm<true, false, true><<<dim3(FFN_DIM / 128, NTOK / 64), 128, 0, stream>>>(
        y, W1T + (size_t)l * FFN_DIM * D_MODEL, b1 + (size_t)l * FFN_DIM,
        nullptr, h1, NTOK, FFN_DIM, D_MODEL);
    k_gemm<false, true, false><<<dim3(D_MODEL / 128, NTOK / 64), 128, 0, stream>>>(
        h1, W2T + (size_t)l * D_MODEL * FFN_DIM, b2 + (size_t)l * D_MODEL,
        x, x, NTOK, D_MODEL, FFN_DIM);
  }

  // ---- final LN + logits ----
  k_layernorm<<<NTOK, 256, 0, stream>>>(x, lnfg, lnfb, y);
  k_gemm<false, false, false><<<dim3(VOCAB / 128, NTOK / 64), 128, 0, stream>>>(
      y, WoutT, bout, nullptr, out, NTOK, VOCAB, D_MODEL);
}
